// GT_33200097198755
// MI455X (gfx1250) — matrix operand order repaired and checked by reading, not compiled, not run on hardware
//
#include <hip/hip_runtime.h>
#include <math.h>

#define N_NODES 50000
#define N_EDGES 800000
#define LAYERS  2
#define HID     128
#define HEADS   8
#define NGRAPH  256

typedef __attribute__((ext_vector_type(16))) __bf16 v16bf;
typedef __attribute__((ext_vector_type(8)))  __bf16 v8bf;
typedef __attribute__((ext_vector_type(8)))  float  v8f;

__device__ __forceinline__ float geluf(float x) {
    return 0.5f * x * (1.0f + erff(x * 0.70710678118654752f));
}
__device__ __forceinline__ unsigned short f32_to_bf16_bits(float f) {
    unsigned int u = __float_as_uint(f);
    u += 0x7FFFu + ((u >> 16) & 1u);   // round-to-nearest-even
    return (unsigned short)(u >> 16);
}
__device__ __forceinline__ float wave_sum32(float v) {
#pragma unroll
    for (int m = 1; m < 32; m <<= 1) v += __shfl_xor(v, m, 32);
    return v;
}
// float atomic-max via sign-split int trick (init memory with a plain negative float)
__device__ __forceinline__ void atomicMaxF32(float* addr, float val) {
    if (val >= 0.0f) atomicMax((int*)addr, __float_as_int(val));
    else             atomicMin((unsigned int*)addr, __float_as_uint(val));
}

// -------------------------------------------------------------------------
__global__ void k_fill_f32(float* p, float v, int n) {
    int i = blockIdx.x * 256 + threadIdx.x;
    if (i < n) p[i] = v;
}

__global__ void k_node_encode(const int* __restrict__ attr,
                              const float* __restrict__ atom_emb,
                              float* __restrict__ x) {
    int n = blockIdx.x, c = threadIdx.x;
    float s = 0.f;
#pragma unroll
    for (int f = 0; f < 4; ++f) {
        int a = attr[n * 4 + f];
        s += atom_emb[((f << 6) + a) * HID + c];
    }
    x[(size_t)n * HID + c] = s;
}

__global__ void k_convert_bf16(const float* __restrict__ in,
                               unsigned short* __restrict__ out, int n, int doGelu) {
    int i = blockIdx.x * 256 + threadIdx.x;
    if (i < n) {
        float v = in[i];
        if (doGelu) v = geluf(v);
        out[i] = f32_to_bf16_bits(v);
    }
}

// C[R x 128] = A[R x 128] @ B[128 x 128]  (tiny relational tables, fp32)
__global__ void k_small_matmul(const float* __restrict__ A, const float* __restrict__ B,
                               float* __restrict__ C, int R) {
    int r = blockIdx.x, c = threadIdx.x;
    if (r >= R) return;
    float s = 0.f;
    for (int k = 0; k < HID; ++k) s += A[r * HID + k] * B[k * HID + c];
    C[r * HID + c] = s;
}

// -------------------------------------------------------------------------
// WMMA bf16 GEMM: C[M x Nc] = epilogue(A[M x K] @ B[K x Nc] + bias [+ residual])
// Nc = NT*16. Block = 256 threads = 8 waves; each wave owns a 16-row strip.
// B staged to LDS in 16x16x32 fragment order (ISA 7.12.2 16-bit layouts).
template <int NT>
__global__ void k_gemm_wmma(const __bf16* __restrict__ A, const __bf16* __restrict__ B,
                            const float* __restrict__ bias, const float* __restrict__ residual,
                            float* __restrict__ C, int M, int K, int actGelu) {
    extern __shared__ __bf16 Bs[];
    const int Nc   = NT * 16;
    const int tid  = threadIdx.x;
    const int lane = tid & 31;
    const int wave = tid >> 5;
    const int ksteps = K >> 5;

    // stage B (K x Nc row-major) into fragment order: [ks][ct][lane][16]
    const int total = K * Nc;
    for (int i = tid; i < total; i += 256) {
        int e  = i & 15;
        int ln = (i >> 4) & 31;
        int ct = (i >> 9) % NT;
        int ks = i / (512 * NT);
        int kidx = ks * 32 + ((e >> 3) << 4) + ((ln >> 4) << 3) + (e & 7);   // lanes 0-15: K {0..7,16..23}; 16-31: K {8..15,24..31}
        int col  = ct * 16 + (ln & 15);
        Bs[i] = B[kidx * Nc + col];
    }
    __syncthreads();

    const int m0 = blockIdx.x * 128 + wave * 16;
    int arow = m0 + (lane & 15);
    if (arow >= M) arow = M - 1;                      // clamp; OOB rows never stored
    const __bf16* Arow = A + (size_t)arow * K;
    const int kofs = (lane >= 16) ? 8 : 0;            // A frag: K {0..7,16..23} / {8..15,24..31}

    v8f acc[NT] = {};

    for (int ks = 0; ks < ksteps; ++ks) {
        const int kb = ks * 32 + kofs;
        v8bf lo = *(const v8bf*)(Arow + kb);
        v8bf hi = *(const v8bf*)(Arow + kb + 16);
        v16bf a;
#pragma unroll
        for (int i = 0; i < 8; ++i) { a[i] = lo[i]; a[i + 8] = hi[i]; }
#pragma unroll
        for (int t = 0; t < NT; ++t) {
            v16bf bfrag = *(const v16bf*)(Bs + (((size_t)ks * NT + t) * 32 + lane) * 16);
            acc[t] = __builtin_amdgcn_wmma_f32_16x16x32_bf16(
                false, a, false, bfrag, (short)0, acc[t], false, false);
        }
    }

    // C/D layout: VGPR r -> row m0 + r (+8 for lanes 16-31), col = lane&15
    const int rbase = m0 + ((lane >= 16) ? 8 : 0);
    const int coll  = lane & 15;
#pragma unroll
    for (int t = 0; t < NT; ++t) {
        int col = t * 16 + coll;
        float bv = bias ? bias[col] : 0.0f;
#pragma unroll
        for (int r = 0; r < 8; ++r) {
            int row = rbase + r;
            if (row < M) {
                float v = acc[t][r] + bv;
                if (residual) v += residual[(size_t)row * Nc + col];
                if (actGelu) v = geluf(v);
                C[(size_t)row * Nc + col] = v;
            }
        }
    }
}

// -------------------------------------------------------------------------
// Edge phase: one wave32 per edge, 4 floats per lane, 4 lanes per head.
__global__ void k_edge_scores(const int* __restrict__ ei, const int* __restrict__ sd,
                              const int* __restrict__ sp,
                              const float* __restrict__ XQ, const float* __restrict__ XK,
                              const float* __restrict__ DKt, const float* __restrict__ PKt,
                              float* __restrict__ scores, float* __restrict__ nmax) {
    int e = blockIdx.x * 8 + (threadIdx.x >> 5);
    if (e >= N_EDGES) return;
    int lane = threadIdx.x & 31;
    int src = ei[e], tgt = ei[N_EDGES + e];
    int d = sd[e], p = sp[e];
    int c4 = lane * 4;
    const float4 q  = *(const float4*)(XQ + (size_t)tgt * HID + c4);
    float4 k        = *(const float4*)(XK + (size_t)src * HID + c4);
    const float4 dk = *(const float4*)(DKt + d * HID + c4);
    const float4 pk = *(const float4*)(PKt + p * HID + c4);
    k.x += dk.x + pk.x; k.y += dk.y + pk.y; k.z += dk.z + pk.z; k.w += dk.w + pk.w;
    float s = q.x * k.x + q.y * k.y + q.z * k.z + q.w * k.w;
    s += __shfl_xor(s, 1, 32);
    s += __shfl_xor(s, 2, 32);
    if ((lane & 3) == 0) {
        int h = lane >> 2;
        float sc = s * 0.25f;                 // 1/sqrt(DK), DK=16
        scores[(size_t)e * HEADS + h] = sc;
        atomicMaxF32(&nmax[tgt * HEADS + h], sc);
    }
}

__global__ void k_edge_exp(const int* __restrict__ ei, float* __restrict__ scores,
                           const float* __restrict__ nmax, float* __restrict__ nsum) {
    int i = blockIdx.x * 256 + threadIdx.x;
    if (i >= N_EDGES * HEADS) return;
    int e = i >> 3, h = i & 7;
    int tgt = ei[N_EDGES + e];
    float ex = expf(scores[i] - nmax[tgt * HEADS + h]);
    scores[i] = ex;
    atomicAdd(&nsum[tgt * HEADS + h], ex);
}

__global__ void k_edge_message(const int* __restrict__ ei, const int* __restrict__ sd,
                               const int* __restrict__ sp,
                               const float* __restrict__ XV, const float* __restrict__ DVt,
                               const float* __restrict__ PVt,
                               const float* __restrict__ scores, const float* __restrict__ nsum,
                               float* __restrict__ aggr) {
    int e = blockIdx.x * 8 + (threadIdx.x >> 5);
    if (e >= N_EDGES) return;
    int lane = threadIdx.x & 31;
    int src = ei[e], tgt = ei[N_EDGES + e];
    int d = sd[e], p = sp[e];
    int h = lane >> 2;
    float att = scores[(size_t)e * HEADS + h] / (nsum[tgt * HEADS + h] + 1e-16f);
    int c4 = lane * 4;
    const float4 v  = *(const float4*)(XV + (size_t)src * HID + c4);
    const float4 dv = *(const float4*)(DVt + d * HID + c4);
    const float4 pv = *(const float4*)(PVt + p * HID + c4);
    float* ap = aggr + (size_t)tgt * HID + c4;
    atomicAdd(ap + 0, (v.x + dv.x + pv.x) * att);
    atomicAdd(ap + 1, (v.y + dv.y + pv.y) * att);
    atomicAdd(ap + 2, (v.z + dv.z + pv.z) * att);
    atomicAdd(ap + 3, (v.w + dv.w + pv.w) * att);
}

__global__ void k_layernorm(float* __restrict__ x, const float* __restrict__ g,
                            const float* __restrict__ b) {
    int n = blockIdx.x * 8 + (threadIdx.x >> 5);
    if (n >= N_NODES) return;
    int lane = threadIdx.x & 31;
    int c = lane * 4;
    float4 v = *(float4*)(x + (size_t)n * HID + c);
    float mean = wave_sum32(v.x + v.y + v.z + v.w) * (1.0f / HID);
    float dx = v.x - mean, dy = v.y - mean, dz = v.z - mean, dw = v.w - mean;
    float var = wave_sum32(dx * dx + dy * dy + dz * dz + dw * dw) * (1.0f / HID);
    float r = rsqrtf(var + 1e-5f);
    float4 o;
    o.x = dx * r * g[c + 0] + b[c + 0];
    o.y = dy * r * g[c + 1] + b[c + 1];
    o.z = dz * r * g[c + 2] + b[c + 2];
    o.w = dw * r * g[c + 3] + b[c + 3];
    *(float4*)(x + (size_t)n * HID + c) = o;
}

__global__ void k_pool_scatter(const int* __restrict__ batch, const float* __restrict__ x,
                               float* __restrict__ pooled, float* __restrict__ cnt) {
    int n = blockIdx.x * 8 + (threadIdx.x >> 5);
    if (n >= N_NODES) return;
    int lane = threadIdx.x & 31;
    int g = batch[n];
    int c = lane * 4;
    const float4 v = *(const float4*)(x + (size_t)n * HID + c);
    float* pp = pooled + (size_t)g * HID + c;
    atomicAdd(pp + 0, v.x); atomicAdd(pp + 1, v.y);
    atomicAdd(pp + 2, v.z); atomicAdd(pp + 3, v.w);
    if (lane == 0) atomicAdd(&cnt[g], 1.0f);
}

__global__ void k_readout(const float* __restrict__ pooled, const float* __restrict__ cnt,
                          const float* __restrict__ W_out, const float* __restrict__ b_out,
                          float* __restrict__ out) {
    int g = blockIdx.x;
    int lane = threadIdx.x;
    int c = lane * 4;
    const float4 v = *(const float4*)(pooled + (size_t)g * HID + c);
    const float4 w = *(const float4*)(W_out + c);
    float s = wave_sum32(v.x * w.x + v.y * w.y + v.z * w.z + v.w * w.w);
    if (lane == 0) {
        float cn = cnt[g]; if (cn < 1.0f) cn = 1.0f;
        out[g] = s / cn + b_out[0];
    }
}

// -------------------------------------------------------------------------
extern "C" void kernel_launch(void* const* d_in, const int* in_sizes, int n_in,
                              void* d_out, int out_size, void* d_ws, size_t ws_size,
                              hipStream_t stream) {
    (void)in_sizes; (void)n_in; (void)out_size; (void)ws_size;

    const int*   node_attr = (const int*)d_in[0];
    const int*   batch_idx = (const int*)d_in[1];
    const int*   edge_idx  = (const int*)d_in[2];
    const int*   strat_d   = (const int*)d_in[3];
    const int*   strat_p   = (const int*)d_in[4];
    const float* atom_emb  = (const float*)d_in[5];
    const float* dist_emb  = (const float*)d_in[6];
    const float* path_emb  = (const float*)d_in[7];
    const float* Wq = (const float*)d_in[8];  const float* bq = (const float*)d_in[9];
    const float* Wk = (const float*)d_in[10]; const float* bk = (const float*)d_in[11];
    const float* Wv = (const float*)d_in[12]; const float* bv = (const float*)d_in[13];
    const float* Wa = (const float*)d_in[14]; const float* ba = (const float*)d_in[15];
    const float* ln1g = (const float*)d_in[16]; const float* ln1b = (const float*)d_in[17];
    const float* Wmid = (const float*)d_in[18]; const float* bmid = (const float*)d_in[19];
    const float* Wo  = (const float*)d_in[20]; const float* bo  = (const float*)d_in[21];
    const float* ln2g = (const float*)d_in[22]; const float* ln2b = (const float*)d_in[23];
    const float* W_out = (const float*)d_in[24]; const float* b_out = (const float*)d_in[25];
    float* out = (float*)d_out;

    char* ws = (char*)d_ws;
    size_t off = 0;
    auto alloc = [&](size_t bytes) -> void* {
        void* p = ws + off;
        off += (bytes + 255) & ~(size_t)255;
        return p;
    };
    float* x    = (float*)alloc((size_t)N_NODES * HID * 4);
    float* XQ   = (float*)alloc((size_t)N_NODES * HID * 4);
    float* XK   = (float*)alloc((size_t)N_NODES * HID * 4);
    float* XV   = (float*)alloc((size_t)N_NODES * HID * 4);
    float* aggr = (float*)alloc((size_t)N_NODES * HID * 4);
    float* hbuf = (float*)alloc((size_t)N_NODES * HID * 4);
    float* midb = (float*)alloc((size_t)N_NODES * 256 * 4);
    unsigned short* abf = (unsigned short*)alloc((size_t)N_NODES * 256 * 2);
    unsigned short* wbf = (unsigned short*)alloc((size_t)256 * 128 * 2);
    float* scores = (float*)alloc((size_t)N_EDGES * HEADS * 4);
    float* nmax = (float*)alloc((size_t)N_NODES * HEADS * 4);
    float* nsum = (float*)alloc((size_t)N_NODES * HEADS * 4);
    float* DKt = (float*)alloc(32 * HID * 4);
    float* PKt = (float*)alloc(16 * HID * 4);
    float* DVt = (float*)alloc(32 * HID * 4);
    float* PVt = (float*)alloc(16 * HID * 4);
    float* pooled = (float*)alloc((size_t)NGRAPH * HID * 4);
    float* cnt    = (float*)alloc((size_t)NGRAPH * 4);

    const int gemm_grid = (N_NODES + 127) / 128;
    const int nNH  = N_NODES * HID;
    const int nN8  = N_NODES * HEADS;
    const int eBlk = (N_EDGES + 7) / 8;
    auto cdiv = [](int a, int b) { return (a + b - 1) / b; };

    k_node_encode<<<N_NODES, HID, 0, stream>>>(node_attr, atom_emb, x);

    for (int l = 0; l < LAYERS; ++l) {
        const float* Wq_l = Wq + (size_t)l * HID * HID;  const float* bq_l = bq + l * HID;
        const float* Wk_l = Wk + (size_t)l * HID * HID;  const float* bk_l = bk + l * HID;
        const float* Wv_l = Wv + (size_t)l * HID * HID;  const float* bv_l = bv + l * HID;
        const float* Wa_l = Wa + (size_t)l * HID * HID;  const float* ba_l = ba + l * HID;
        const float* Wm_l = Wmid + (size_t)l * HID * 256; const float* bm_l = bmid + l * 256;
        const float* Wo_l = Wo + (size_t)l * 256 * HID;  const float* bo_l = bo + l * HID;
        const float* de_l = dist_emb + (size_t)l * 32 * HID;
        const float* pe_l = path_emb + (size_t)l * 16 * HID;

        // node projections: XQ/XK/XV = x@W + b  (bf16 WMMA, fp32 accum)
        k_convert_bf16<<<cdiv(nNH, 256), 256, 0, stream>>>(x, abf, nNH, 0);
        k_convert_bf16<<<cdiv(HID * HID, 256), 256, 0, stream>>>(Wq_l, wbf, HID * HID, 0);
        k_gemm_wmma<8><<<gemm_grid, 256, HID * HID * 2, stream>>>(
            (const __bf16*)abf, (const __bf16*)wbf, bq_l, nullptr, XQ, N_NODES, HID, 0);
        k_convert_bf16<<<cdiv(HID * HID, 256), 256, 0, stream>>>(Wk_l, wbf, HID * HID, 0);
        k_gemm_wmma<8><<<gemm_grid, 256, HID * HID * 2, stream>>>(
            (const __bf16*)abf, (const __bf16*)wbf, bk_l, nullptr, XK, N_NODES, HID, 0);
        k_convert_bf16<<<cdiv(HID * HID, 256), 256, 0, stream>>>(Wv_l, wbf, HID * HID, 0);
        k_gemm_wmma<8><<<gemm_grid, 256, HID * HID * 2, stream>>>(
            (const __bf16*)abf, (const __bf16*)wbf, bv_l, nullptr, XV, N_NODES, HID, 0);

        // relational tables (linearity: (x+d+p)@W = x@W + d@W + p@W)
        k_small_matmul<<<32, HID, 0, stream>>>(de_l, Wk_l, DKt, 32);
        k_small_matmul<<<16, HID, 0, stream>>>(pe_l, Wk_l, PKt, 16);
        k_small_matmul<<<32, HID, 0, stream>>>(de_l, Wv_l, DVt, 32);
        k_small_matmul<<<16, HID, 0, stream>>>(pe_l, Wv_l, PVt, 16);

        // segment softmax + scatter aggregation
        k_fill_f32<<<cdiv(nN8, 256), 256, 0, stream>>>(nmax, -3.0e38f, nN8);
        k_fill_f32<<<cdiv(nN8, 256), 256, 0, stream>>>(nsum, 0.0f, nN8);
        k_fill_f32<<<cdiv(nNH, 256), 256, 0, stream>>>(aggr, 0.0f, nNH);
        k_edge_scores<<<eBlk, 256, 0, stream>>>(edge_idx, strat_d, strat_p,
                                                XQ, XK, DKt, PKt, scores, nmax);
        k_edge_exp<<<cdiv(N_EDGES * HEADS, 256), 256, 0, stream>>>(edge_idx, scores, nmax, nsum);
        k_edge_message<<<eBlk, 256, 0, stream>>>(edge_idx, strat_d, strat_p,
                                                 XV, DVt, PVt, scores, nsum, aggr);

        // h = LN(gelu(aggr)@Wa + ba + x)
        k_convert_bf16<<<cdiv(nNH, 256), 256, 0, stream>>>(aggr, abf, nNH, 1);
        k_convert_bf16<<<cdiv(HID * HID, 256), 256, 0, stream>>>(Wa_l, wbf, HID * HID, 0);
        k_gemm_wmma<8><<<gemm_grid, 256, HID * HID * 2, stream>>>(
            (const __bf16*)abf, (const __bf16*)wbf, ba_l, x, hbuf, N_NODES, HID, 0);
        k_layernorm<<<cdiv(N_NODES, 8), 256, 0, stream>>>(hbuf, ln1g + l * HID, ln1b + l * HID);

        // mid = gelu(h@Wmid + bmid)   (Nc = 256)
        k_convert_bf16<<<cdiv(nNH, 256), 256, 0, stream>>>(hbuf, abf, nNH, 0);
        k_convert_bf16<<<cdiv(HID * 256, 256), 256, 0, stream>>>(Wm_l, wbf, HID * 256, 0);
        k_gemm_wmma<16><<<gemm_grid, 256, HID * 256 * 2, stream>>>(
            (const __bf16*)abf, (const __bf16*)wbf, bm_l, nullptr, midb, N_NODES, HID, 1);

        // x = LN(mid@Wo + bo + h)   (K = 256)
        k_convert_bf16<<<cdiv(N_NODES * 256, 256), 256, 0, stream>>>(midb, abf, N_NODES * 256, 0);
        k_convert_bf16<<<cdiv(256 * HID, 256), 256, 0, stream>>>(Wo_l, wbf, 256 * HID, 0);
        k_gemm_wmma<8><<<gemm_grid, 256, 256 * HID * 2, stream>>>(
            (const __bf16*)abf, (const __bf16*)wbf, bo_l, hbuf, x, N_NODES, 256, 0);
        k_layernorm<<<cdiv(N_NODES, 8), 256, 0, stream>>>(x, ln2g + l * HID, ln2b + l * HID);
    }

    // mean pool per graph + readout
    k_fill_f32<<<cdiv(NGRAPH * HID, 256), 256, 0, stream>>>(pooled, 0.0f, NGRAPH * HID);
    k_fill_f32<<<cdiv(NGRAPH, 256), 256, 0, stream>>>(cnt, 0.0f, NGRAPH);
    k_pool_scatter<<<cdiv(N_NODES, 8), 256, 0, stream>>>(batch_idx, x, pooled, cnt);
    k_readout<<<NGRAPH, 32, 0, stream>>>(pooled, cnt, W_out, b_out, out);
}
